// BottomLevelDecoderRNN_32658931318789
// MI455X (gfx1250) — compile-verified, matrix operand order checked by reading, not run on hardware
//
#include <hip/hip_runtime.h>

// ---------------------------------------------------------------------------
// BottomLevelDecoderRNN for MI455X (gfx1250, wave32, WMMA bf16, async-LDS,
// double-buffered K-loop, manually 2x-unrolled with constant buffer ids).
//
// Per time step (3 fused kernels):
//   1) lstm_gates_fused(X1cur, Wcat1): g = X1 @ Wcat1^T + b1, LSTM cell
//      update fused in epilogue -> c1, h1 (bf16 into X1nxt.h and X2cur.h1)
//   2) lstm_gates_fused(X2cur, Wcat2): -> c2, h2 (bf16 into X2nxt.h2)
//   3) out_gemm_fused: out = h2 @ W_out^T + b_out -> d_out[b,tg,:] (fp32)
//      and prev (bf16) into X1nxt.prev
// Segment init GEMM fuses tanh + h1/h2/c1/c2 scatter into its epilogue.
// X1 = [embed | prev | h1] (256x2048 bf16), X2 = [h1 | h2]; both ping-pong.
//
// K-loop protocol (per chunk, single barrier):
//   s_wait_asynccnt 0 -> own async writes landed
//   __syncthreads()   -> all waves' writes visible AND all waves done
//                        reading the buffer we are about to overwrite
//   stage next chunk (async, overlaps compute) ; wmma on current buffer
// ---------------------------------------------------------------------------

typedef __attribute__((ext_vector_type(16))) __bf16 v16bf;
typedef __attribute__((ext_vector_type(8)))  float  v8f;

#define N_SEG 16
#define BATCH 256
#define COND  512
#define HID   1024
#define OUTD  512
#define STEPS 16                  // length(=32)/2, fixed by reference setup
#define TSTEPS (N_SEG * STEPS)

#define BM 64
#define BN 64
#define BK 32
#define LDT 40                    // LDS row stride in bf16 (80B, 16B aligned)

#define USE_ASYNC_COPY 1

union Frag { uint4 u[2]; v16bf v; };

__device__ __forceinline__ unsigned short f2bf(float f) {
    union { float f; unsigned int u; } x; x.f = f;
    unsigned int r = x.u + 0x7FFFu + ((x.u >> 16) & 1u);   // RNE
    return (unsigned short)(r >> 16);
}

__device__ __forceinline__ float sigmoidf_(float x) {
    return 1.0f / (1.0f + __expf(-x));
}

// stage one uint4 (8 bf16) per lane from global into LDS
__device__ __forceinline__ void ld_lds_b128(void* lds, const void* g) {
#if USE_ASYNC_COPY
    unsigned l = (unsigned)(unsigned long long)lds;         // LDS byte offset
    unsigned long long ga = (unsigned long long)g;
    asm volatile("global_load_async_to_lds_b128 %0, %1, off"
                 :: "v"(l), "v"(ga) : "memory");
#else
    *(uint4*)lds = *(const uint4*)g;
#endif
}

__device__ __forceinline__ void lds_copy_fence() {
#if USE_ASYNC_COPY
    asm volatile("s_wait_asynccnt 0x0" ::: "memory");
#endif
}

// ---------------------------------------------------------------------------
// Fused gates GEMM + LSTM cell update.
// X: [256 x 2048] bf16 (lda 2048). W: [4096 x 2048] bf16 gate-major (i,f,g,o).
// Block computes batch-rows [blockM,+64) x gate-cols [blockN,+64) in ALL FOUR
// slabs, so each lane owns (i,f,g,o) of its elements in registers.
// grid = (1024/BN, 256/BM) = (16,4); 8 waves: wm in {0,1} x 32 rows,
// wn in {0..3} x 16 cols; per wave 2 M-tiles x 4 slabs = 8 accumulators.
// ---------------------------------------------------------------------------
__global__ __launch_bounds__(256)
void lstm_gates_fused(const unsigned short* __restrict__ X,
                      const unsigned short* __restrict__ W,
                      const float* __restrict__ bias,
                      float* __restrict__ cst,
                      unsigned short* __restrict__ dH1,
                      unsigned short* __restrict__ dH2)
{
    __shared__ unsigned short sA[2][BM * LDT];
    __shared__ unsigned short sB[2][4 * BN * LDT];

    const int tid  = threadIdx.x;
    const int wave = tid >> 5, lane = tid & 31;
    const int wm = wave & 1, wn = wave >> 1;
    const int blockN = blockIdx.x * BN;          // col within one gate slab
    const int blockM = blockIdx.y * BM;

    const int lr = tid >> 2;                     // cooperative-load row 0..63
    const int lc = (tid & 3) * 8;                // K offset 0,8,16,24

    const unsigned short* Ag = X + (size_t)(blockM + lr) * 2048 + lc;
    const unsigned short* Wg = W + (size_t)(blockN + lr) * 2048 + lc;
    const int sAo = lr * LDT + lc;

    v8f acc[4][2] = {};

    const int half = (lane >> 4) * 8;
    const int row  = lane & 15;

    auto stage = [&](int buf, int k) {
        ld_lds_b128(&sA[buf][sAo], Ag + k);
#pragma unroll
        for (int s = 0; s < 4; ++s)
            ld_lds_b128(&sB[buf][(s * BN + lr) * LDT + lc],
                        Wg + (size_t)s * HID * 2048 + k);
    };

    auto compute = [&](int buf) {
        Frag fa[2], fb[4];
#pragma unroll
        for (int mt = 0; mt < 2; ++mt) {
            const unsigned short* p = &sA[buf][(wm * 32 + mt * 16 + row) * LDT + half];
            fa[mt].u[0] = *(const uint4*)(p);
            fa[mt].u[1] = *(const uint4*)(p + 16);
        }
#pragma unroll
        for (int s = 0; s < 4; ++s) {
            const unsigned short* p = &sB[buf][(s * BN + wn * 16 + row) * LDT + half];
            fb[s].u[0] = *(const uint4*)(p);
            fb[s].u[1] = *(const uint4*)(p + 16);
        }
#pragma unroll
        for (int s = 0; s < 4; ++s)
#pragma unroll
            for (int mt = 0; mt < 2; ++mt)
                acc[s][mt] = __builtin_amdgcn_wmma_f32_16x16x32_bf16(
                    false, fa[mt].v, false, fb[s].v, (short)0, acc[s][mt],
                    false, false);
    };

    const int NIT = 2048 / BK;                   // 64 chunks (even)
    stage(0, 0);
#pragma clang loop unroll(disable)
    for (int i = 0; i < NIT - 2; i += 2) {
        lds_copy_fence(); __syncthreads();
        stage(1, (i + 1) * BK);
        compute(0);
        lds_copy_fence(); __syncthreads();
        stage(0, (i + 2) * BK);
        compute(1);
    }
    lds_copy_fence(); __syncthreads();
    stage(1, (NIT - 1) * BK);
    compute(0);
    lds_copy_fence(); __syncthreads();
    compute(1);

    // fused LSTM cell update (PyTorch gate order i,f,g,o)
    const int col = blockN + wn * 16 + (lane & 15);
    const float bI = bias[col];
    const float bF = bias[col + 1024];
    const float bG = bias[col + 2048];
    const float bO = bias[col + 3072];
    const int rb = blockM + wm * 32 + ((lane >> 4) ? 8 : 0);
#pragma unroll
    for (int mt = 0; mt < 2; ++mt) {
#pragma unroll
        for (int v = 0; v < 8; ++v) {
            const int r = rb + mt * 16 + v;
            float gi = acc[0][mt][v] + bI;
            float gf = acc[1][mt][v] + bF;
            float gg = acc[2][mt][v] + bG;
            float go = acc[3][mt][v] + bO;
            float cc = cst[(size_t)r * HID + col];
            float cn = sigmoidf_(gf) * cc + sigmoidf_(gi) * tanhf(gg);
            float h  = sigmoidf_(go) * tanhf(cn);
            cst[(size_t)r * HID + col] = cn;
            unsigned short hb = f2bf(h);
            dH1[(size_t)r * 2048 + col] = hb;
            if (dH2) dH2[(size_t)r * 2048 + col] = hb;
        }
    }
}

// ---------------------------------------------------------------------------
// Segment init: t = tanh(E @ W_init^T + b_init); scatter h1/h2/c1/c2.
// E: [256 x 512] bf16 (lda 512), Wi: [4096 x 512]. grid (64,4).
// ---------------------------------------------------------------------------
__global__ __launch_bounds__(256)
void init_gemm_fused(const unsigned short* __restrict__ E,
                     const unsigned short* __restrict__ Wi,
                     const float* __restrict__ bias,
                     unsigned short* __restrict__ X1c,
                     unsigned short* __restrict__ X2c,
                     float* __restrict__ c1, float* __restrict__ c2)
{
    __shared__ unsigned short sA[2][BM * LDT];
    __shared__ unsigned short sB[2][BN * LDT];

    const int tid  = threadIdx.x;
    const int wave = tid >> 5, lane = tid & 31;
    const int wm = wave & 1, wn = wave >> 1;
    const int blockN = blockIdx.x * BN;
    const int blockM = blockIdx.y * BM;
    const int lr = tid >> 2;
    const int lc = (tid & 3) * 8;

    const unsigned short* Ag = E  + (size_t)(blockM + lr) * COND + lc;
    const unsigned short* Wg = Wi + (size_t)(blockN + lr) * COND + lc;
    const int so = lr * LDT + lc;

    v8f acc0 = {}, acc1 = {};
    const int half = (lane >> 4) * 8;
    const int row  = lane & 15;

    auto stage = [&](int buf, int k) {
        ld_lds_b128(&sA[buf][so], Ag + k);
        ld_lds_b128(&sB[buf][so], Wg + k);
    };
    auto compute = [&](int buf) {
        Frag fa0, fa1, fb;
        const unsigned short* pa0 = &sA[buf][(wm * 32 + row) * LDT + half];
        fa0.u[0] = *(const uint4*)(pa0);
        fa0.u[1] = *(const uint4*)(pa0 + 16);
        const unsigned short* pa1 = pa0 + 16 * LDT;
        fa1.u[0] = *(const uint4*)(pa1);
        fa1.u[1] = *(const uint4*)(pa1 + 16);
        const unsigned short* pb = &sB[buf][(wn * 16 + row) * LDT + half];
        fb.u[0] = *(const uint4*)(pb);
        fb.u[1] = *(const uint4*)(pb + 16);
        acc0 = __builtin_amdgcn_wmma_f32_16x16x32_bf16(
            false, fa0.v, false, fb.v, (short)0, acc0, false, false);
        acc1 = __builtin_amdgcn_wmma_f32_16x16x32_bf16(
            false, fa1.v, false, fb.v, (short)0, acc1, false, false);
    };

    const int NIT = COND / BK;                   // 16 chunks (even)
    stage(0, 0);
#pragma clang loop unroll(disable)
    for (int i = 0; i < NIT - 2; i += 2) {
        lds_copy_fence(); __syncthreads();
        stage(1, (i + 1) * BK);
        compute(0);
        lds_copy_fence(); __syncthreads();
        stage(0, (i + 2) * BK);
        compute(1);
    }
    lds_copy_fence(); __syncthreads();
    stage(1, (NIT - 1) * BK);
    compute(0);
    lds_copy_fence(); __syncthreads();
    compute(1);

    const int col  = blockN + wn * 16 + (lane & 15);
    const float bv = bias[col];
    const int rb   = blockM + wm * 32 + ((lane >> 4) ? 8 : 0);
#pragma unroll
    for (int mt = 0; mt < 2; ++mt) {
#pragma unroll
        for (int v = 0; v < 8; ++v) {
            const int r = rb + mt * 16 + v;
            float t = tanhf((mt ? acc1[v] : acc0[v]) + bv);
            if (col < HID) {
                X1c[(size_t)r * 2048 + 1024 + col] = f2bf(t);
            } else if (col < 2 * HID) {
                X2c[(size_t)r * 2048 + 1024 + (col - HID)] = f2bf(t);
            } else if (col < 3 * HID) {
                c1[(size_t)r * HID + (col - 2 * HID)] = t;
            } else {
                c2[(size_t)r * HID + (col - 3 * HID)] = t;
            }
        }
    }
}

// ---------------------------------------------------------------------------
// Output projection: out = H2 @ Wo^T + b_out; writes fp32 straight into
// d_out[b, tg, :] (ldc = TSTEPS*OUTD) and bf16 prev into X1nxt.prev.
// H2: rows at stride 2048 (X2nxt + 1024). Wo: [512 x 1024]. grid (8,4).
// ---------------------------------------------------------------------------
__global__ __launch_bounds__(256)
void out_gemm_fused(const unsigned short* __restrict__ H2,
                    const unsigned short* __restrict__ Wo,
                    const float* __restrict__ bias,
                    float* __restrict__ outBase,
                    unsigned short* __restrict__ prevBase)
{
    __shared__ unsigned short sA[2][BM * LDT];
    __shared__ unsigned short sB[2][BN * LDT];

    const int tid  = threadIdx.x;
    const int wave = tid >> 5, lane = tid & 31;
    const int wm = wave & 1, wn = wave >> 1;
    const int blockN = blockIdx.x * BN;
    const int blockM = blockIdx.y * BM;
    const int lr = tid >> 2;
    const int lc = (tid & 3) * 8;

    const unsigned short* Ag = H2 + (size_t)(blockM + lr) * 2048 + lc;
    const unsigned short* Wg = Wo + (size_t)(blockN + lr) * HID + lc;
    const int so = lr * LDT + lc;

    v8f acc0 = {}, acc1 = {};
    const int half = (lane >> 4) * 8;
    const int row  = lane & 15;

    auto stage = [&](int buf, int k) {
        ld_lds_b128(&sA[buf][so], Ag + k);
        ld_lds_b128(&sB[buf][so], Wg + k);
    };
    auto compute = [&](int buf) {
        Frag fa0, fa1, fb;
        const unsigned short* pa0 = &sA[buf][(wm * 32 + row) * LDT + half];
        fa0.u[0] = *(const uint4*)(pa0);
        fa0.u[1] = *(const uint4*)(pa0 + 16);
        const unsigned short* pa1 = pa0 + 16 * LDT;
        fa1.u[0] = *(const uint4*)(pa1);
        fa1.u[1] = *(const uint4*)(pa1 + 16);
        const unsigned short* pb = &sB[buf][(wn * 16 + row) * LDT + half];
        fb.u[0] = *(const uint4*)(pb);
        fb.u[1] = *(const uint4*)(pb + 16);
        acc0 = __builtin_amdgcn_wmma_f32_16x16x32_bf16(
            false, fa0.v, false, fb.v, (short)0, acc0, false, false);
        acc1 = __builtin_amdgcn_wmma_f32_16x16x32_bf16(
            false, fa1.v, false, fb.v, (short)0, acc1, false, false);
    };

    const int NIT = HID / BK;                    // 32 chunks (even)
    stage(0, 0);
#pragma clang loop unroll(disable)
    for (int i = 0; i < NIT - 2; i += 2) {
        lds_copy_fence(); __syncthreads();
        stage(1, (i + 1) * BK);
        compute(0);
        lds_copy_fence(); __syncthreads();
        stage(0, (i + 2) * BK);
        compute(1);
    }
    lds_copy_fence(); __syncthreads();
    stage(1, (NIT - 1) * BK);
    compute(0);
    lds_copy_fence(); __syncthreads();
    compute(1);

    const int col  = blockN + wn * 16 + (lane & 15);
    const float bv = bias[col];
    const int rb   = blockM + wm * 32 + ((lane >> 4) ? 8 : 0);
#pragma unroll
    for (int mt = 0; mt < 2; ++mt) {
#pragma unroll
        for (int v = 0; v < 8; ++v) {
            const int r = rb + mt * 16 + v;
            float val = (mt ? acc1[v] : acc0[v]) + bv;
            outBase[(size_t)r * (TSTEPS * OUTD) + col] = val;
            prevBase[(size_t)r * 2048 + col] = f2bf(val);
        }
    }
}

// ---- setup / small kernels -------------------------------------------------

__global__ __launch_bounds__(256)
void concat_convert_kernel(const float* __restrict__ Wih,
                           const float* __restrict__ Whh,
                           unsigned short* __restrict__ Wcat)
{
    long long idx = (long long)blockIdx.x * 256 + threadIdx.x;  // 4096*2048
    int n = (int)(idx >> 11);
    int k = (int)(idx & 2047);
    float v = (k < HID) ? Wih[(long long)n * HID + k]
                        : Whh[(long long)n * HID + (k - HID)];
    Wcat[idx] = f2bf(v);
}

__global__ __launch_bounds__(256)
void convert_bf16_kernel(const float* __restrict__ src,
                         unsigned short* __restrict__ dst, int n)
{
    int idx = blockIdx.x * 256 + threadIdx.x;
    if (idx < n) dst[idx] = f2bf(src[idx]);
}

__global__ __launch_bounds__(256)
void bias_sum_kernel(const float* __restrict__ a, const float* __restrict__ b,
                     float* __restrict__ dst, int n)
{
    int idx = blockIdx.x * 256 + threadIdx.x;
    if (idx < n) dst[idx] = a[idx] + b[idx];
}

__global__ __launch_bounds__(256)
void zero_prev_kernel(unsigned short* __restrict__ X1)
{
    int idx = blockIdx.x * 256 + threadIdx.x;   // 256*512
    int b = idx >> 9, o = idx & 511;
    X1[b * 2048 + 512 + o] = 0;                 // bf16 zero
}

__global__ __launch_bounds__(256)
void embed_copy_kernel(const unsigned short* __restrict__ cseg,
                       unsigned short* __restrict__ Xa,
                       unsigned short* __restrict__ Xb)
{
    int idx = blockIdx.x * 256 + threadIdx.x;   // 256*512
    int b = idx >> 9, o = idx & 511;
    unsigned short v = cseg[idx];
    Xa[b * 2048 + o] = v;
    Xb[b * 2048 + o] = v;
}

__global__ void ratio_kernel(const int* __restrict__ epoch,
                             const int* __restrict__ k,
                             float* __restrict__ dst)
{
    float kk = (float)(*k);
    float e  = (float)(*epoch);
    *dst = kk / (kk + expf(e / kk));
}

// ---------------------------------------------------------------------------

extern "C" void kernel_launch(void* const* d_in, const int* in_sizes, int n_in,
                              void* d_out, int out_size, void* d_ws, size_t ws_size,
                              hipStream_t stream)
{
    const float* c_in   = (const float*)d_in[0];
    const float* W_init = (const float*)d_in[1];
    const float* b_init = (const float*)d_in[2];
    const float* W_ih1  = (const float*)d_in[3];
    const float* W_hh1  = (const float*)d_in[4];
    const float* b_ih1  = (const float*)d_in[5];
    const float* b_hh1  = (const float*)d_in[6];
    const float* W_ih2  = (const float*)d_in[7];
    const float* W_hh2  = (const float*)d_in[8];
    const float* b_ih2  = (const float*)d_in[9];
    const float* b_hh2  = (const float*)d_in[10];
    const float* W_out  = (const float*)d_in[11];
    const float* b_out  = (const float*)d_in[12];
    const int*   epoch  = (const int*)d_in[14];
    const int*   kval   = (const int*)d_in[15];

    float* out = (float*)d_out;

    char* p = (char*)d_ws;
    auto alloc = [&](size_t bytes) {
        void* r = (void*)p;
        p += (bytes + 255) & ~(size_t)255;
        return r;
    };
    unsigned short* Wcat1   = (unsigned short*)alloc((size_t)4096 * 2048 * 2);
    unsigned short* Wcat2   = (unsigned short*)alloc((size_t)4096 * 2048 * 2);
    unsigned short* Winit_b = (unsigned short*)alloc((size_t)4096 * 512 * 2);
    unsigned short* Wout_b  = (unsigned short*)alloc((size_t)512 * 1024 * 2);
    unsigned short* c_bf    = (unsigned short*)alloc((size_t)N_SEG * BATCH * COND * 2);
    unsigned short* X1buf[2], * X2buf[2];
    X1buf[0] = (unsigned short*)alloc((size_t)BATCH * 2048 * 2);
    X1buf[1] = (unsigned short*)alloc((size_t)BATCH * 2048 * 2);
    X2buf[0] = (unsigned short*)alloc((size_t)BATCH * 2048 * 2);
    X2buf[1] = (unsigned short*)alloc((size_t)BATCH * 2048 * 2);
    float* c1 = (float*)alloc((size_t)BATCH * HID * 4);
    float* c2 = (float*)alloc((size_t)BATCH * HID * 4);
    float* b1 = (float*)alloc(4096 * 4);
    float* b2 = (float*)alloc(4096 * 4);

    // one-time weight prep
    concat_convert_kernel<<<(4096 * 2048) / 256, 256, 0, stream>>>(W_ih1, W_hh1, Wcat1);
    concat_convert_kernel<<<(4096 * 2048) / 256, 256, 0, stream>>>(W_ih2, W_hh2, Wcat2);
    convert_bf16_kernel<<<(4096 * 512) / 256, 256, 0, stream>>>(W_init, Winit_b, 4096 * 512);
    convert_bf16_kernel<<<(512 * 1024) / 256, 256, 0, stream>>>(W_out, Wout_b, 512 * 1024);
    convert_bf16_kernel<<<(N_SEG * BATCH * COND) / 256, 256, 0, stream>>>(
        c_in, c_bf, N_SEG * BATCH * COND);
    bias_sum_kernel<<<16, 256, 0, stream>>>(b_ih1, b_hh1, b1, 4096);
    bias_sum_kernel<<<16, 256, 0, stream>>>(b_ih2, b_hh2, b2, 4096);
    zero_prev_kernel<<<(BATCH * OUTD) / 256, 256, 0, stream>>>(X1buf[0]);

    const dim3 blk(256);
    const dim3 grid_gates(HID / BN, BATCH / BM);    // (16, 4)
    const dim3 grid_init(4096 / BN, BATCH / BM);    // (64, 4)
    const dim3 grid_out(OUTD / BN, BATCH / BM);     // (8, 4)

    int cur = 0;
    for (int s = 0; s < N_SEG; ++s) {
        const unsigned short* cseg = c_bf + (size_t)s * BATCH * COND;

        // embed slot of both ping-pong X1 buffers + fresh LSTM state
        embed_copy_kernel<<<(BATCH * COND) / 256, blk, 0, stream>>>(
            cseg, X1buf[0], X1buf[1]);
        init_gemm_fused<<<grid_init, blk, 0, stream>>>(
            cseg, Winit_b, b_init, X1buf[cur], X2buf[cur], c1, c2);

        for (int t = 0; t < STEPS; ++t) {
            const int tg  = s * STEPS + t;
            const int nxt = cur ^ 1;

            // layer 1: reads X1cur, writes h1 -> X1nxt.h & X2cur.h1
            lstm_gates_fused<<<grid_gates, blk, 0, stream>>>(
                X1buf[cur], Wcat1, b1, c1, X1buf[nxt] + 1024, X2buf[cur]);

            // layer 2: reads X2cur, writes h2 -> X2nxt.h2
            lstm_gates_fused<<<grid_gates, blk, 0, stream>>>(
                X2buf[cur], Wcat2, b2, c2, X2buf[nxt] + 1024,
                (unsigned short*)nullptr);

            // projection: d_out[b,tg,:] + prev -> X1nxt.prev
            out_gemm_fused<<<grid_out, blk, 0, stream>>>(
                X2buf[nxt] + 1024, Wout_b, b_out,
                out + (long long)tg * OUTD, X1buf[nxt] + 512);

            cur = nxt;
        }
    }

    ratio_kernel<<<1, 1, 0, stream>>>(epoch, kval, out + (out_size - 1));
}
